// DeformConv_45999099740837
// MI455X (gfx1250) — compile-verified, matrix operand order checked by reading, not run on hardware
//
#include <hip/hip_runtime.h>
#include <hip/hip_bf16.h>

typedef __attribute__((ext_vector_type(16))) __bf16 v16bf;
typedef __attribute__((ext_vector_type(8)))  float  v8f;

union ABfrag { v16bf v; uint4 q[2]; };

#define HH   128
#define WW   128
#define CIN  128
#define COUT 256
#define KDIM 1152          // 9 * CIN, K index = n*128 + c (reordered!)
#define NBATCH 4
#define HWSZ 16384         // HH*WW
#define NPIX 65536         // NBATCH*HWSZ
#define PIXT 64            // pixels per workgroup in both GEMM kernels
#define KSTEPS (KDIM / 32)
#define EPS  1e-5f
#define LEAK 0.1f

// ---- workspace layout (bytes) ----
// [0, 4096)           float stats: sum[256] | sumsq[256] | scale[256] | shift[256]
// [4096, 593920)      bf16 main weights,   256 x 1152, K = n*128+c
// [594944, 668672)    bf16 offset weights,  32 x 1152, K = n*128+c (rows 18..31 zero)
// [1<<20, +4718592)   fp32 offsets, NPIX x 18
// [6<<20, +33554432)  fp32 x transposed to NHWC: (b, h, w, c)
#define WS_WBF_OFF   4096u
#define WS_PWBF_OFF  594944u
#define WS_OFFS_OFF  (1u << 20)
#define WS_XT_OFF    (6u << 20)

__device__ __forceinline__ unsigned short f2bf(float f) {
    unsigned int u = __float_as_uint(f);
    unsigned int r = u + 0x7fffu + ((u >> 16) & 1u);   // round-to-nearest-even
    return (unsigned short)(r >> 16);
}
__device__ __forceinline__ unsigned int pk2bf(float a, float b) {
    return (unsigned int)f2bf(a) | ((unsigned int)f2bf(b) << 16);
}

// ------------------------------------------------------------------
__global__ void zero_stats(float* stats) {
    int i = blockIdx.x * blockDim.x + threadIdx.x;
    if (i < 512) stats[i] = 0.0f;
}

// ------------------------------------------------------------------
// NCHW -> NHWC tiled transpose through LDS (coalesced both sides)
__global__ void transpose_x(const float* __restrict__ x, float* __restrict__ xt) {
    __shared__ float tile[32][33];
    int b = blockIdx.z;
    int hw0 = blockIdx.x * 32, c0 = blockIdx.y * 32;
    int tx = threadIdx.x & 31, ty = threadIdx.x >> 5;      // 32 x 8
#pragma unroll
    for (int i = 0; i < 32; i += 8)
        tile[ty + i][tx] = x[((size_t)(b * CIN + c0 + ty + i) << 14) + hw0 + tx];
    __syncthreads();
#pragma unroll
    for (int i = 0; i < 32; i += 8)
        xt[(((size_t)(b << 14) + hw0 + ty + i) << 7) + c0 + tx] = tile[tx][ty + i];
}

// ------------------------------------------------------------------
// main conv weights: (o, c, n) fp32 -> (o, K=n*128+c) bf16
__global__ void convert_w(const float* __restrict__ w, unsigned short* __restrict__ wbf) {
    int i = blockIdx.x * blockDim.x + threadIdx.x;         // COUT*KDIM
    if (i < COUT * KDIM) {
        int o = i / KDIM, k = i % KDIM, n = k >> 7, c = k & 127;
        wbf[i] = f2bf(w[(o * CIN + c) * 9 + n]);
    }
}
// offset conv weights: (18, c, n) fp32 -> (32, K=n*128+c) bf16, zero padded rows
__global__ void convert_pw(const float* __restrict__ pw, unsigned short* __restrict__ pwbf) {
    int i = blockIdx.x * blockDim.x + threadIdx.x;         // 32*KDIM
    if (i < 32 * KDIM) {
        int o = i / KDIM, k = i % KDIM, n = k >> 7, c = k & 127;
        pwbf[i] = (o < 18) ? f2bf(pw[(o * CIN + c) * 9 + n]) : (unsigned short)0;
    }
}

// ------------------------------------------------------------------
// Offset conv as WMMA GEMM: im2col A (64 pix x 1152) in LDS, N = 32 (18 live)
__global__ void offset_gemm(const float* __restrict__ xt, const unsigned short* __restrict__ pwbf,
                            const float* __restrict__ pb, float* __restrict__ off) {
    extern __shared__ unsigned short ldsA[];               // PIXT * KDIM bf16 = 147456 B
    const int base = blockIdx.x * PIXT;

    // ---- phase 1: im2col (contiguous in c thanks to NHWC) ----
    for (int pair = threadIdx.x; pair < PIXT * 9; pair += 256) {
        int lp = pair / 9, n = pair % 9;
        int pix = base + lp;
        int b = pix >> 14, hw = pix & (HWSZ - 1);
        int h = hw >> 7, w = hw & (WW - 1);
        int hh = h + n / 3 - 1, ww2 = w + n % 3 - 1;
        bool ok = (hh >= 0) & (hh < HH) & (ww2 >= 0) & (ww2 < WW);
        uint2* arow = (uint2*)(ldsA + lp * KDIM + (n << 7));
        if (ok) {
            const float4* src = (const float4*)(xt + (((size_t)(b << 14) + hh * WW + ww2) << 7));
#pragma unroll 4
            for (int c4 = 0; c4 < 32; ++c4) {
                float4 v = src[c4];
                uint2 p; p.x = pk2bf(v.x, v.y); p.y = pk2bf(v.z, v.w);
                arow[c4] = p;
            }
        } else {
            uint2 z; z.x = 0u; z.y = 0u;
            for (int c4 = 0; c4 < 32; ++c4) arow[c4] = z;
        }
    }
    __syncthreads();

    // ---- phase 2: 4 M-tiles x 2 N-tiles, one 16x16 tile per wave ----
    const int lane = threadIdx.x & 31, wv = threadIdx.x >> 5;
    const int mt = wv >> 1, nt = wv & 1;
    const int half = lane >> 4, lr = lane & 15;

    v8f acc = {};
    const unsigned short* arow = ldsA + (mt * 16 + lr) * KDIM;
    const unsigned short* brow = pwbf + (size_t)(nt * 16 + lr) * KDIM;
    for (int kk = 0; kk < KSTEPS; ++kk) {
        const int kb = kk * 32;
        ABfrag a, bf;
        const uint4* ap = (const uint4*)(arow + kb);
        const uint4* bp = (const uint4*)(brow + kb);
        a.q[0]  = ap[half];     a.q[1]  = ap[2 + half];
        bf.q[0] = bp[half];     bf.q[1] = bp[2 + half];
        acc = __builtin_amdgcn_wmma_f32_16x16x32_bf16(
            false, a.v, false, bf.v, (short)0, acc, false, false);
    }

    int nch = nt * 16 + lr;
    if (nch < 18) {
        float bias = pb[nch];
#pragma unroll
        for (int r = 0; r < 8; ++r) {
            int pix = base + mt * 16 + r + half * 8;
            off[(size_t)pix * 18 + nch] = acc[r] + bias;
        }
    }
}

// ------------------------------------------------------------------
// Fused deformable sampling (bf16 A in LDS) + WMMA GEMM + BN partial sums
__global__ void deform_gemm(const float* __restrict__ xt, const float* __restrict__ off,
                            const unsigned short* __restrict__ wbf,
                            float* __restrict__ out, float* __restrict__ stats) {
    extern __shared__ unsigned short ldsA[];               // PIXT * KDIM bf16 = 147456 B
    const int base = blockIdx.x * PIXT;

    // ---- phase 1: bilinear sampling, branch-free (invalid corner -> weight 0) ----
    for (int pair = threadIdx.x; pair < PIXT * 9; pair += 256) {
        int lp = pair / 9, n = pair % 9;
        int pix = base + lp;
        int b = pix >> 14, hw = pix & (HWSZ - 1);
        int h = hw >> 7, w = hw & (WW - 1);

        float ox = off[(size_t)pix * 18 + n];
        float oy = off[(size_t)pix * 18 + 9 + n];
        float px = (float)(h + 1) + (float)(n / 3 - 1) + ox;    // padded coords, Hp=130
        float py = (float)(w + 1) + (float)(n % 3 - 1) + oy;
        float fx = floorf(px), fy = floorf(py);
        float qltx = fminf(fmaxf(fx, 0.f), 129.f);
        float qlty = fminf(fmaxf(fy, 0.f), 129.f);
        float qrbx = fminf(fmaxf(fx + 1.f, 0.f), 129.f);
        float qrby = fminf(fmaxf(fy + 1.f, 0.f), 129.f);
        float pcx  = fminf(fmaxf(px, 0.f), 129.f);
        float pcy  = fminf(fmaxf(py, 0.f), 129.f);
        float glt = (1.f + (qltx - pcx)) * (1.f + (qlty - pcy));
        float grb = (1.f - (qrbx - pcx)) * (1.f - (qrby - pcy));
        float glb = (1.f + (qltx - pcx)) * (1.f - (qrby - pcy));
        float grt = (1.f - (qrbx - pcx)) * (1.f + (qlty - pcy));

        int ix0 = (int)qltx - 1, iy0 = (int)qlty - 1;           // unpadded coords
        int ix1 = (int)qrbx - 1, iy1 = (int)qrby - 1;
        bool vx0 = (ix0 >= 0) & (ix0 < HH), vy0 = (iy0 >= 0) & (iy0 < WW);
        bool vx1 = (ix1 >= 0) & (ix1 < HH), vy1 = (iy1 >= 0) & (iy1 < WW);
        bool vlt = vx0 & vy0, vrb = vx1 & vy1, vlb = vx0 & vy1, vrt = vx1 & vy0;
        if (!vlt) glt = 0.f;
        if (!vrb) grb = 0.f;
        if (!vlb) glb = 0.f;
        if (!vrt) grt = 0.f;
        const float4* xtb = (const float4*)(xt + ((size_t)b << 21));  // b*HWSZ*128
        const float4* r_lt = xtb + ((size_t)(vlt ? ix0 * WW + iy0 : 0) << 5);
        const float4* r_rb = xtb + ((size_t)(vrb ? ix1 * WW + iy1 : 0) << 5);
        const float4* r_lb = xtb + ((size_t)(vlb ? ix0 * WW + iy1 : 0) << 5);
        const float4* r_rt = xtb + ((size_t)(vrt ? ix1 * WW + iy0 : 0) << 5);

        uint2* arow = (uint2*)(ldsA + lp * KDIM + (n << 7));
#pragma unroll 4
        for (int c4 = 0; c4 < 32; ++c4) {
            float4 a = r_lt[c4], bb = r_rb[c4], cc = r_lb[c4], dd = r_rt[c4];
            float s0 = glt * a.x + grb * bb.x + glb * cc.x + grt * dd.x;
            float s1 = glt * a.y + grb * bb.y + glb * cc.y + grt * dd.y;
            float s2 = glt * a.z + grb * bb.z + glb * cc.z + grt * dd.z;
            float s3 = glt * a.w + grb * bb.w + glb * cc.w + grt * dd.w;
            uint2 p; p.x = pk2bf(s0, s1); p.y = pk2bf(s2, s3);
            arow[c4] = p;
        }
    }
    __syncthreads();

    // ---- phase 2: WMMA GEMM (64 x 1152) x (1152 x 256) ----
    // wave -> 2 M-tiles x 4 N-tiles: 8 WMMA / K-step, each B frag feeds 2 WMMAs
    const int lane = threadIdx.x & 31, wv = threadIdx.x >> 5;
    const int mtp = wv & 1;              // M pair: rows [32*mtp, 32*mtp+32)
    const int ng  = wv >> 1;             // N group of 4 tiles
    const int half = lane >> 4, lr = lane & 15;

    v8f acc[2][4] = {};
    const unsigned short* arow0 = ldsA + (mtp * 32 + lr) * KDIM;
    const unsigned short* arow1 = arow0 + 16 * KDIM;

    for (int kk = 0; kk < KSTEPS; ++kk) {
        const int kb = kk * 32;
        ABfrag a0, a1;
        const uint4* ap0 = (const uint4*)(arow0 + kb);
        const uint4* ap1 = (const uint4*)(arow1 + kb);
        a0.q[0] = ap0[half]; a0.q[1] = ap0[2 + half];
        a1.q[0] = ap1[half]; a1.q[1] = ap1[2 + half];
#pragma unroll
        for (int j = 0; j < 4; ++j) {
            int nch = (ng * 4 + j) * 16 + lr;
            const uint4* bp = (const uint4*)(wbf + (size_t)nch * KDIM + kb);
            ABfrag bf;
            bf.q[0] = bp[half];
            bf.q[1] = bp[2 + half];
            if (kk + 1 < KSTEPS)
                __builtin_prefetch((const void*)(bp + 4), 0, 1);   // next K-block of this row
            acc[0][j] = __builtin_amdgcn_wmma_f32_16x16x32_bf16(
                false, a0.v, false, bf.v, (short)0, acc[0][j], false, false);
            acc[1][j] = __builtin_amdgcn_wmma_f32_16x16x32_bf16(
                false, a1.v, false, bf.v, (short)0, acc[1][j], false, false);
        }
    }

    // ---- epilogue: raw output (b,o,h,w) + per-channel BN partial sums ----
#pragma unroll
    for (int mi = 0; mi < 2; ++mi) {
#pragma unroll
        for (int j = 0; j < 4; ++j) {
            int nch = (ng * 4 + j) * 16 + lr;
            float s = 0.f, s2 = 0.f;
#pragma unroll
            for (int r = 0; r < 8; ++r) {
                int m = mtp * 32 + mi * 16 + r + half * 8;
                int pix = base + m;
                int b = pix >> 14, hw = pix & (HWSZ - 1);
                float val = acc[mi][j][r];
                out[(((size_t)b * COUT + nch) << 14) + hw] = val;
                s += val;
                s2 = fmaf(val, val, s2);
            }
            atomicAdd(&stats[nch], s);
            atomicAdd(&stats[COUT + nch], s2);
        }
    }
}

// ------------------------------------------------------------------
__global__ void finalize_stats(const float* __restrict__ gamma, const float* __restrict__ beta,
                               float* stats) {
    int c = threadIdx.x;
    if (c < COUT) {
        const float inv_n = 1.0f / (float)NPIX;
        float mean = stats[c] * inv_n;
        float var  = stats[COUT + c] * inv_n - mean * mean;
        float sc   = gamma[c] * rsqrtf(var + EPS);
        stats[2 * COUT + c] = sc;
        stats[3 * COUT + c] = beta[c] - mean * sc;
    }
}

// ------------------------------------------------------------------
__global__ void bn_lrelu(float* __restrict__ out, const float* __restrict__ stats) {
    int i = blockIdx.x * 256 + threadIdx.x;                 // float4 index
    if (i < (NBATCH * COUT * HWSZ) / 4) {
        int ch = (i >> 12) & (COUT - 1);
        float sc = stats[2 * COUT + ch];
        float sh = stats[3 * COUT + ch];
        float4 v = ((float4*)out)[i];
        float r0 = fmaf(v.x, sc, sh); v.x = r0 >= 0.f ? r0 : LEAK * r0;
        float r1 = fmaf(v.y, sc, sh); v.y = r1 >= 0.f ? r1 : LEAK * r1;
        float r2 = fmaf(v.z, sc, sh); v.z = r2 >= 0.f ? r2 : LEAK * r2;
        float r3 = fmaf(v.w, sc, sh); v.w = r3 >= 0.f ? r3 : LEAK * r3;
        ((float4*)out)[i] = v;
    }
}

// ------------------------------------------------------------------
extern "C" void kernel_launch(void* const* d_in, const int* in_sizes, int n_in,
                              void* d_out, int out_size, void* d_ws, size_t ws_size,
                              hipStream_t stream) {
    const float* x      = (const float*)d_in[0];
    const float* p_w    = (const float*)d_in[1];
    const float* p_b    = (const float*)d_in[2];
    const float* w_conv = (const float*)d_in[3];
    const float* gamma  = (const float*)d_in[4];
    const float* beta   = (const float*)d_in[5];
    float* out = (float*)d_out;

    float*          stats = (float*)d_ws;
    unsigned short* wbf   = (unsigned short*)((char*)d_ws + WS_WBF_OFF);
    unsigned short* pwbf  = (unsigned short*)((char*)d_ws + WS_PWBF_OFF);
    float*          off   = (float*)((char*)d_ws + WS_OFFS_OFF);
    float*          xt    = (float*)((char*)d_ws + WS_XT_OFF);

    zero_stats<<<2, 256, 0, stream>>>(stats);
    transpose_x<<<dim3(HWSZ / 32, CIN / 32, NBATCH), 256, 0, stream>>>(x, xt);
    convert_w<<<(COUT * KDIM + 255) / 256, 256, 0, stream>>>(w_conv, wbf);
    convert_pw<<<(32 * KDIM + 255) / 256, 256, 0, stream>>>(p_w, pwbf);
    offset_gemm<<<NPIX / PIXT, 256, PIXT * KDIM * sizeof(unsigned short), stream>>>(
        xt, pwbf, p_b, off);
    deform_gemm<<<NPIX / PIXT, 256, PIXT * KDIM * sizeof(unsigned short), stream>>>(
        xt, off, wbf, out, stats);
    finalize_stats<<<1, 256, 0, stream>>>(gamma, beta, stats);
    bn_lrelu<<<(NBATCH * COUT * HWSZ / 4 + 255) / 256, 256, 0, stream>>>(out, stats);
}
